// DeepCapsNet_59330678227613
// MI455X (gfx1250) — matrix operand order repaired and checked by reading, not compiled, not run on hardware
//
#include <hip/hip_runtime.h>
#include <hip/hip_bf16.h>

typedef __attribute__((ext_vector_type(16))) _Float16 v16h;
typedef __attribute__((ext_vector_type(8)))  _Float16 v8h;
typedef __attribute__((ext_vector_type(8)))  float    v8f;

#define BN_EPS 1e-5f
#define SQ_EPS 1e-20f

// ---------------------------------------------------------------------------
// Kernel 1: conv1 3x3 stride2 (Cin=1) + BN + ReLU.  NCHW f32 in -> NHWC f16.
// h1 layout: (512, 19, 19, 128) so conv2 K-chunks are contiguous channels.
// ---------------------------------------------------------------------------
__global__ void k_conv1(const float* __restrict__ x,      // (512,1,39,39)
                        const float* __restrict__ w,      // (128,1,3,3)
                        const float* __restrict__ bias,
                        const float* __restrict__ g, const float* __restrict__ bb,
                        const float* __restrict__ mu, const float* __restrict__ vr,
                        _Float16* __restrict__ h1)
{
    int idx = blockIdx.x * blockDim.x + threadIdx.x;
    const int total = 512 * 19 * 19 * 128;
    if (idx >= total) return;
    int c  = idx & 127;
    int t  = idx >> 7;
    int ox = t % 19; t /= 19;
    int oy = t % 19; t /= 19;
    int b  = t;
    const float* xb = x + (size_t)b * 39 * 39;
    const float* wc = w + c * 9;
    int iy = 2 * oy, ix = 2 * ox;
    float acc = 0.f;
#pragma unroll
    for (int ky = 0; ky < 3; ++ky)
#pragma unroll
        for (int kx = 0; kx < 3; ++kx)
            acc = fmaf(xb[(iy + ky) * 39 + (ix + kx)], wc[ky * 3 + kx], acc);
    float inv = g[c] * rsqrtf(vr[c] + BN_EPS);
    float val = (acc + bias[c]) * inv + (bb[c] - mu[c] * inv);
    h1[idx] = (_Float16)fmaxf(val, 0.f);
}

// ---------------------------------------------------------------------------
// Kernel 2: conv2 weights (256,128,3,3) f32 -> f16 with K = (ky*3+kx)*128 + c
// ---------------------------------------------------------------------------
__global__ void k_w2cvt(const float* __restrict__ w, _Float16* __restrict__ wh)
{
    int idx = blockIdx.x * blockDim.x + threadIdx.x;   // mo*1152 + s*128 + c
    if (idx >= 256 * 1152) return;
    int c  = idx & 127;
    int s  = (idx >> 7) % 9;
    int mo = idx / 1152;
    wh[idx] = (_Float16)w[(mo * 128 + c) * 9 + s];
}

// ---------------------------------------------------------------------------
// Kernel 3: conv2 as implicit GEMM with V_WMMA_F32_16X16X32_F16.
//   M = Cout = 256, N = B*81 = 41472, K = 1152 (36 chunks of 32).
//   4 waves/block; wave w owns M rows [blockIdx.y*64 + w*16, +16), N tile
//   [blockIdx.x*16, +16).  Fused bias + BN + ReLU epilogue, f16 NCHW out.
// ---------------------------------------------------------------------------
__global__ void __launch_bounds__(128)
k_conv2_wmma(const _Float16* __restrict__ h1,   // (512,19,19,128) NHWC f16
             const _Float16* __restrict__ wh,   // (256,1152) f16
             const float* __restrict__ bias,
             const float* __restrict__ g, const float* __restrict__ bb,
             const float* __restrict__ mu, const float* __restrict__ vr,
             _Float16* __restrict__ h2)         // (512,256,81) NCHW f16
{
    const int lane = threadIdx.x & 31;
    const int wave = threadIdx.x >> 5;
    const int half = lane >> 4;       // 0: lanes 0-15, 1: lanes 16-31
    const int l16  = lane & 15;
    const int m0   = blockIdx.y * 64 + wave * 16;
    const int col  = blockIdx.x * 16 + l16;      // < 41472 always
    const int b    = col / 81;
    const int rem  = col % 81;
    const int oy   = rem / 9, ox = rem % 9;

    const _Float16* bbase = h1 + (size_t)b * 19 * 19 * 128;
    const _Float16* arow  = wh + (size_t)(m0 + l16) * 1152 + half * 16;

    v8f acc = {};
    for (int kk = 0; kk < 36; ++kk) {
        const int k0 = kk * 32;
        const int s  = k0 >> 7;          // spatial tap (128 % 32 == 0)
        const int c0 = k0 & 127;
        const int ky = s / 3, kx = s % 3;
        // B fragment: col = l16, K = k0 + half*16 + [0..15]  (32B aligned)
        const _Float16* bp =
            bbase + (size_t)((2 * oy + ky) * 19 + (2 * ox + kx)) * 128 + c0 + half * 16;
        v16h bfrag = *(const v16h*)bp;
        // A fragment: row = m0+l16, K = k0 + half*16 + [0..15]
        v8h alo = *(const v8h*)(arow + k0);
        v8h ahi = *(const v8h*)(arow + k0 + 8);
        v16h afrag;
#pragma unroll
        for (int e = 0; e < 8; ++e) { afrag[e] = alo[e]; afrag[8 + e] = ahi[e]; }
        acc = __builtin_amdgcn_wmma_f32_16x16x32_f16(
            /*neg_a=*/false, afrag, /*neg_b=*/false, bfrag,
            /*c_mod=*/(short)0, acc, /*reuse_a=*/false, /*reuse_b=*/false);
    }
    // D layout: VGPR r holds M = m0 + half*8 + r at N = col.
#pragma unroll
    for (int r = 0; r < 8; ++r) {
        int M = m0 + half * 8 + r;
        float inv = g[M] * rsqrtf(vr[M] + BN_EPS);
        float val = (acc[r] + bias[M]) * inv + (bb[M] - mu[M] * inv);
        h2[((size_t)b * 256 + M) * 81 + rem] = (_Float16)fmaxf(val, 0.f);
    }
}

// ---------------------------------------------------------------------------
// Kernel 4: depthwise 9x9 conv (-> (512,256,1,1)) + reshape (512,32,8) + squash
// One thread per (b, capsule i): 8 channels x 81 taps.
// ---------------------------------------------------------------------------
__global__ void k_conv3_squash(const _Float16* __restrict__ h2, // (512,256,81)
                               const float* __restrict__ w,     // (256,81)
                               const float* __restrict__ bias,
                               float* __restrict__ u0)          // (512,32,8)
{
    int idx = blockIdx.x * blockDim.x + threadIdx.x;
    if (idx >= 512 * 32) return;
    int i = idx & 31;
    int b = idx >> 5;
    float vals[8];
    float ss = 0.f;
#pragma unroll
    for (int q = 0; q < 8; ++q) {
        int c = i * 8 + q;
        const _Float16* hp = h2 + ((size_t)b * 256 + c) * 81;
        const float* wp = w + c * 81;
        float acc = bias[c];
        for (int p = 0; p < 81; ++p) acc = fmaf((float)hp[p], wp[p], acc);
        vals[q] = acc;
        ss = fmaf(acc, acc, ss);
    }
    float norm = sqrtf(ss);
    float f = (1.f - 1.f / (expf(norm) + SQ_EPS)) / (norm + SQ_EPS);
#pragma unroll
    for (int q = 0; q < 8; ++q) u0[(size_t)idx * 8 + q] = vals[q] * f;
}

// ---------------------------------------------------------------------------
// Kernel 5: all three FC-caps layers, one block per batch sample, LDS-resident.
// Uses A_sum[h,k] = sum_l U_hat[h,k,l] * S[k,l], S = sum_i U_hat[i,k,l].
// ---------------------------------------------------------------------------
__device__ __forceinline__ void caps_layer(int n_h, int d_h,
                                           const float* __restrict__ W,
                                           float* Ul, float* Uh, float* S,
                                           float* As, float* Cs, float* Vh,
                                           float* outp, int tid)
{
    const int n_l = 32, d_l = 8;
    const float inv_scale = 0.35355339059327373f;   // 1/sqrt(8) == 1/ATTN_SCALE
    const int T2 = n_h * d_h;
    // U_hat[i,k,l]
    for (int t = tid; t < n_l * T2; t += 256) {
        int l = t % d_h;
        int k = (t / d_h) % n_h;
        int i = t / T2;
        const float* wp = W + ((size_t)(i * n_h + k) * d_l) * d_h + l;
        const float* up = Ul + i * d_l;
        float acc = 0.f;
#pragma unroll
        for (int j = 0; j < 8; ++j) acc = fmaf(up[j], wp[j * d_h], acc);
        Uh[t] = acc;
    }
    __syncthreads();
    // S[k,l] = sum_i U_hat[i,k,l]
    for (int t = tid; t < T2; t += 256) {
        float acc = 0.f;
        for (int i = 0; i < n_l; ++i) acc += Uh[i * T2 + t];
        S[t] = acc;
    }
    __syncthreads();
    // A_sum[i,k] (pre-scaled by 1/sqrt(8))
    for (int t = tid; t < n_l * n_h; t += 256) {
        int k = t % n_h;
        int i = t / n_h;
        const float* up = Uh + (i * n_h + k) * d_h;
        const float* sp = S + k * d_h;
        float acc = 0.f;
        for (int l = 0; l < d_h; ++l) acc = fmaf(up[l], sp[l], acc);
        As[t] = acc * inv_scale;
    }
    __syncthreads();
    // softmax over k per row i
    if (tid < n_l) {
        float mx = -1e30f;
        for (int k = 0; k < n_h; ++k) mx = fmaxf(mx, As[tid * n_h + k]);
        float sum = 0.f;
        for (int k = 0; k < n_h; ++k) {
            float e = expf(As[tid * n_h + k] - mx);
            Cs[tid * n_h + k] = e;
            sum += e;
        }
        float r = 1.f / sum;
        for (int k = 0; k < n_h; ++k) Cs[tid * n_h + k] *= r;
    }
    __syncthreads();
    // U_h[k,l] = sum_i C[i,k] * U_hat[i,k,l]
    for (int t = tid; t < T2; t += 256) {
        int k = t / d_h;
        float acc = 0.f;
        for (int i = 0; i < n_l; ++i) acc = fmaf(Cs[i * n_h + k], Uh[i * T2 + t], acc);
        Vh[t] = acc;
    }
    __syncthreads();
    // squash -> next Ul (d_h==8 for non-final layers) or global out
    if (tid < n_h) {
        float ss = 0.f;
        for (int l = 0; l < d_h; ++l) { float v = Vh[tid * d_h + l]; ss = fmaf(v, v, ss); }
        float norm = sqrtf(ss);
        float f = (1.f - 1.f / (expf(norm) + SQ_EPS)) / (norm + SQ_EPS);
        if (outp) {
            for (int l = 0; l < d_h; ++l) outp[tid * d_h + l] = Vh[tid * d_h + l] * f;
        } else {
            for (int l = 0; l < d_h; ++l) Ul[tid * 8 + l] = Vh[tid * d_h + l] * f;
        }
    }
    __syncthreads();
}

__global__ void __launch_bounds__(256)
k_caps_head(const float* __restrict__ u0,   // (512,32,8)
            const float* __restrict__ W1,   // (32,32,8,8)
            const float* __restrict__ W2,   // (32,32,8,8)
            const float* __restrict__ W3,   // (32,10,8,16)
            float* __restrict__ out)        // (512,10,16)
{
    __shared__ float Ul[256];
    __shared__ float Uh[8192];
    __shared__ float S[256];
    __shared__ float As[1024];
    __shared__ float Cs[1024];
    __shared__ float Vh[256];
    const int tid = threadIdx.x;
    const int b = blockIdx.x;
    Ul[tid] = u0[(size_t)b * 256 + tid];
    __syncthreads();
    caps_layer(32, 8,  W1, Ul, Uh, S, As, Cs, Vh, nullptr,             tid);
    caps_layer(32, 8,  W2, Ul, Uh, S, As, Cs, Vh, nullptr,             tid);
    caps_layer(10, 16, W3, Ul, Uh, S, As, Cs, Vh, out + (size_t)b*160, tid);
}

// ---------------------------------------------------------------------------
extern "C" void kernel_launch(void* const* d_in, const int* in_sizes, int n_in,
                              void* d_out, int out_size, void* d_ws, size_t ws_size,
                              hipStream_t stream)
{
    const float* x       = (const float*)d_in[0];
    const float* conv1_w = (const float*)d_in[1];
    const float* conv1_b = (const float*)d_in[2];
    const float* bn1_g   = (const float*)d_in[3];
    const float* bn1_b   = (const float*)d_in[4];
    const float* bn1_m   = (const float*)d_in[5];
    const float* bn1_v   = (const float*)d_in[6];
    const float* conv2_w = (const float*)d_in[7];
    const float* conv2_b = (const float*)d_in[8];
    const float* bn2_g   = (const float*)d_in[9];
    const float* bn2_b   = (const float*)d_in[10];
    const float* bn2_m   = (const float*)d_in[11];
    const float* bn2_v   = (const float*)d_in[12];
    const float* conv3_w = (const float*)d_in[13];
    const float* conv3_b = (const float*)d_in[14];
    const float* W1      = (const float*)d_in[15];
    const float* W2      = (const float*)d_in[16];
    const float* W3      = (const float*)d_in[17];
    float* out = (float*)d_out;

    // workspace layout (all offsets 256B aligned)
    char* wsb = (char*)d_ws;
    const size_t H1_BYTES = (size_t)512 * 19 * 19 * 128 * 2;   // 47,316,992
    const size_t W2_BYTES = (size_t)256 * 1152 * 2;            //    589,824
    const size_t H2_BYTES = (size_t)512 * 256 * 81 * 2;        // 21,233,664
    _Float16* h1  = (_Float16*)(wsb);
    _Float16* w2h = (_Float16*)(wsb + H1_BYTES);
    _Float16* h2  = (_Float16*)(wsb + H1_BYTES + W2_BYTES);
    float*    u0  = (float*)   (wsb + H1_BYTES + W2_BYTES + H2_BYTES);

    // 1. conv1 + BN + ReLU -> h1 (NHWC f16)
    {
        int total = 512 * 19 * 19 * 128;
        k_conv1<<<(total + 255) / 256, 256, 0, stream>>>(
            x, conv1_w, conv1_b, bn1_g, bn1_b, bn1_m, bn1_v, h1);
    }
    // 2. conv2 weight permute+convert
    {
        int total = 256 * 1152;
        k_w2cvt<<<(total + 255) / 256, 256, 0, stream>>>(conv2_w, w2h);
    }
    // 3. conv2 implicit GEMM (WMMA f16 -> f32) + BN + ReLU -> h2 (NCHW f16)
    {
        dim3 grid(41472 / 16, 256 / 64);   // (2592, 4)
        k_conv2_wmma<<<grid, 128, 0, stream>>>(
            h1, w2h, conv2_b, bn2_g, bn2_b, bn2_m, bn2_v, h2);
    }
    // 4. depthwise conv3 + squash -> u0 (512,32,8) f32
    {
        int total = 512 * 32;
        k_conv3_squash<<<(total + 255) / 256, 256, 0, stream>>>(
            h2, conv3_w, conv3_b, u0);
    }
    // 5. three FC-caps layers -> out (512,10,16)
    k_caps_head<<<512, 256, 0, stream>>>(u0, W1, W2, W3, out);
}
